// FP8QuantizedLinear_40149354283484
// MI455X (gfx1250) — matrix operand order repaired and checked by reading, not compiled, not run on hardware
//
#include <hip/hip_runtime.h>
#include <hip/hip_bf16.h>

typedef __attribute__((ext_vector_type(16))) __bf16 v16bf;
typedef __attribute__((ext_vector_type(8)))  __bf16 v8bf;
typedef __attribute__((ext_vector_type(8)))  float  v8f;
typedef __attribute__((ext_vector_type(4)))  int    v4i;

#if __has_builtin(__builtin_amdgcn_global_load_async_to_lds_b128) && \
    __has_builtin(__builtin_amdgcn_s_wait_asynccnt)
#define USE_ASYNC_LDS 1
typedef __attribute__((address_space(1))) v4i gas_v4i;  // global (addrspace 1) v4i
typedef __attribute__((address_space(3))) v4i las_v4i;  // LDS    (addrspace 3) v4i
#else
#define USE_ASYNC_LDS 0
#endif

namespace {
constexpr int kM = 256;               // B*S
constexpr int kK = 8192;              // DIN
constexpr int kN = 8192;              // DOUT
constexpr int kKStep = 32;            // K per WMMA bf16 op
constexpr int kNK = kK / kKStep;      // 256 K-iterations (even -> clean 2x unroll)
constexpr int kXStride = 40;          // LDS row stride in bf16 elems (80B: 16B-aligned, bank-conflict-free)
constexpr int kXBuf = kM * kXStride;  // 10240 elems (20KB) per LDS buffer
}

__global__ __launch_bounds__(256)
void qlinear_bf16_wmma_kernel(const unsigned short* __restrict__ xg,
                              const int* __restrict__ wq,
                              const float* __restrict__ wscale,
                              const __hip_bfloat16* __restrict__ bias,
                              __hip_bfloat16* __restrict__ outg)
{
  __shared__ __align__(16) unsigned short xs[2 * kXBuf];  // 40 KB, double-buffered x tile

  const int tid   = threadIdx.x;
  const int lane  = tid & 31;
  const int wave  = tid >> 5;
  const int waveM = wave & 3;            // 4 M-groups of 64 rows
  const int waveN = wave >> 2;           // 2 N-groups of 32 cols
  const int nbase = blockIdx.x * 64 + waveN * 32;

  const int nlo = lane & 15;             // column within a 16-wide N tile
  const int kq  = (lane >> 4) << 4;      // B-frag: this lane holds K = kq..kq+15
  const int kla = (lane >> 4) << 3;      // unused; A-frag also holds K = kq..kq+15

  const int* wrow0 = wq + (size_t)(nbase + nlo)      * kK + kq;
  const int* wrow1 = wq + (size_t)(nbase + 16 + nlo) * kK + kq;

  v8f acc[4][2];
#pragma unroll
  for (int mt = 0; mt < 4; ++mt)
#pragma unroll
    for (int nt = 0; nt < 2; ++nt)
      acc[mt][nt] = v8f{0.f, 0.f, 0.f, 0.f, 0.f, 0.f, 0.f, 0.f};

  // --- stage 256x32 bf16 x-tile into LDS (all 256 threads, 4x16B chunks each).
  //     CDNA5 path: async DMA straight into LDS, tracked by ASYNCcnt.
  auto stage_x = [&](int buf, int kbase) {
#pragma unroll
    for (int i = 0; i < 4; ++i) {
      const int linear = tid + 256 * i;     // 1024 chunks = 256 rows x 4 chunks
      const int row = linear >> 2;
      const int col = linear & 3;           // 16B chunk within the row's 64B slab
      const unsigned short* gp = xg + (size_t)row * kK + kbase + col * 8;
      unsigned short* lp = &xs[buf * kXBuf + row * kXStride + col * 8];
#if USE_ASYNC_LDS
      __builtin_amdgcn_global_load_async_to_lds_b128(
          (gas_v4i*)gp, (las_v4i*)lp, 0, 0);
#else
      *(uint4*)lp = *(const uint4*)gp;
#endif
    }
  };

  auto wait_async = [&]() {
#if USE_ASYNC_LDS
    __builtin_amdgcn_s_wait_asynccnt(0);
#endif
  };

  // --- weight fragment loads: 8 x b128 per lane per K-step, non-temporal (read-once stream)
  auto load_w = [&](v4i* wr, int kbase) {
#pragma unroll
    for (int c = 0; c < 4; ++c) {
      wr[c]     = __builtin_nontemporal_load((const v4i*)(wrow0 + kbase + 4 * c));
      wr[4 + c] = __builtin_nontemporal_load((const v4i*)(wrow1 + kbase + 4 * c));
    }
  };

  // --- dequant int32 -> bf16 B fragment (scale folded into epilogue)
  auto cvt_b = [&](const v4i* wr, int nt) {
    v16bf b;
#pragma unroll
    for (int j = 0; j < 16; ++j)
      b[j] = (__bf16)(float)wr[nt * 4 + (j >> 2)][j & 3];
    return b;
  };

  // --- A fragment from LDS: two ds_load_b128 per lane
  auto load_a = [&](int buf, int mt) {
    const int row = waveM * 64 + mt * 16 + nlo;
    const unsigned short* lp = &xs[buf * kXBuf + row * kXStride + kq];
    union { v16bf v; v8bf h[2]; } u;
    u.h[0] = *(const v8bf*)lp;           // K = kq .. kq+7
    u.h[1] = *(const v8bf*)(lp + 8);     // K = kq+8 .. kq+15
    return u.v;
  };

  auto compute = [&](int buf, const v4i* wr) {
    const v16bf b0 = cvt_b(wr, 0);
    const v16bf b1 = cvt_b(wr, 1);
#pragma unroll
    for (int mt = 0; mt < 4; ++mt) {
      const v16bf a = load_a(buf, mt);
      acc[mt][0] = __builtin_amdgcn_wmma_f32_16x16x32_bf16(
          false, a, false, b0, (short)0, acc[mt][0], false, false);
      acc[mt][1] = __builtin_amdgcn_wmma_f32_16x16x32_bf16(
          false, a, false, b1, (short)0, acc[mt][1], false, false);
    }
  };

  v4i wreg[2][8];
  load_w(wreg[0], 0);
  stage_x(0, 0);
  wait_async();
  __syncthreads();

  // K loop manually unrolled by 2: buffer indices are compile-time constants,
  // so there is no wcur/wnxt register rotation and no LDS base-pointer select.
#pragma unroll 1
  for (int ks = 0; ks < kNK; ks += 2) {
    if (ks + 1 < kNK) {                       // prefetch step ks+1 into buf1/wreg1
      stage_x(1, (ks + 1) * kKStep);
      load_w(wreg[1], (ks + 1) * kKStep);
    }
    compute(0, wreg[0]);
    wait_async();
    __syncthreads();

    if (ks + 2 < kNK) {                       // prefetch step ks+2 into buf0/wreg0
      stage_x(0, (ks + 2) * kKStep);
      load_w(wreg[0], (ks + 2) * kKStep);
    }
    compute(1, wreg[1]);
    wait_async();
    __syncthreads();
  }

  // --- epilogue: acc * scale + bias -> bf16
  const float scale = wscale[0];
  const int mrow0 = waveM * 64 + ((lane >> 4) << 3);   // C layout: VGPR r -> M = r (+8 for upper lanes)
#pragma unroll
  for (int nt = 0; nt < 2; ++nt) {
    const int n = nbase + nt * 16 + nlo;
    const float bf = __bfloat162float(bias[n]);
#pragma unroll
    for (int mt = 0; mt < 4; ++mt) {
#pragma unroll
      for (int r = 0; r < 8; ++r) {
        const int m = mrow0 + mt * 16 + r;
        const float v = acc[mt][nt][r] * scale + bf;
        outg[(size_t)m * kN + n] = __float2bfloat16(v);
      }
    }
  }
}

extern "C" void kernel_launch(void* const* d_in, const int* in_sizes, int n_in,
                              void* d_out, int out_size, void* d_ws, size_t ws_size,
                              hipStream_t stream) {
  const unsigned short*  x    = (const unsigned short*)d_in[0];   // bf16 raw
  const int*             wqp  = (const int*)d_in[1];              // int8 values in int32
  const float*           ws   = (const float*)d_in[2];            // scalar scale
  const __hip_bfloat16*  bptr = (const __hip_bfloat16*)d_in[3];   // bias
  __hip_bfloat16*        out  = (__hip_bfloat16*)d_out;

  dim3 grid(kN / 64);     // 128 blocks: each owns 64 output columns, all 256 rows
  dim3 block(256);        // 8 waves (wave32)
  qlinear_bf16_wmma_kernel<<<grid, block, 0, stream>>>(x, wqp, ws, bptr, out);
}
